// ContextAttention_4535485464624
// MI455X (gfx1250) — matrix operand order repaired and checked by reading, not compiled, not run on hardware
//
#include <hip/hip_runtime.h>
#include <hip/hip_bf16.h>
#include <math.h>

// Problem dims (fixed by the reference)
#define B_  128
#define S_  2048
#define D_  512
#define H_  512

// ---------------- WMMA types ----------------
typedef __attribute__((ext_vector_type(16))) __bf16 v16bf;
typedef __attribute__((ext_vector_type(8)))  float  v8f;

__device__ __forceinline__ unsigned short f2bf(float f) {
    unsigned u = __float_as_uint(f);
    unsigned r = 0x7FFFu + ((u >> 16) & 1u);   // round-to-nearest-even
    return (unsigned short)((u + r) >> 16);
}
__device__ __forceinline__ unsigned pack2(float lo, float hi) {
    return (unsigned)f2bf(lo) | ((unsigned)f2bf(hi) << 16);
}

// Branch-free tanh: 1 - 2/(exp(2x)+1).  Saturates correctly at +-1.
__device__ __forceinline__ float fast_tanh(float x) {
    float e = __expf(2.0f * x);
    return 1.0f - 2.0f * __builtin_amdgcn_rcpf(e + 1.0f);
}

union ABFrag { v16bf v; unsigned u[8]; };

// ---------------------------------------------------------------------------
// Convert f32 -> bf16 (used once for W_ctx). 4 elems/thread.
// ---------------------------------------------------------------------------
__global__ __launch_bounds__(256) void to_bf16(
    const float* __restrict__ src, unsigned short* __restrict__ dst)
{
    int i = blockIdx.x * 256 + threadIdx.x;     // float4 index
    float4 x = ((const float4*)src)[i];
    ((uint2*)dst)[i] = make_uint2(pack2(x.x, x.y), pack2(x.z, x.w));
}

// ---------------------------------------------------------------------------
// Y[b,h] = (sum_p X[p][b,:]) . W[h,:] + bias[h] (+ bias2[h])
// X may be split into `xparts` partial buffers of stride `xstride` (summed
// during the LDS stage) -- used to fold the weighted-context reduction in.
// grid (B, 2), block 256
// ---------------------------------------------------------------------------
__global__ __launch_bounds__(256) void linear_bh(
    const float* __restrict__ X, const float* __restrict__ W,
    const float* __restrict__ bias, const float* __restrict__ bias2,
    float* __restrict__ Y, int xparts, size_t xstride)
{
    __shared__ float xs[D_];
    const int b = blockIdx.x;
    const int t = threadIdx.x;
    float x0 = 0.f, x1 = 0.f;
    for (int p = 0; p < xparts; ++p) {
        x0 += X[(size_t)p * xstride + b * D_ + t];
        x1 += X[(size_t)p * xstride + b * D_ + t + 256];
    }
    xs[t]       = x0;
    xs[t + 256] = x1;
    __syncthreads();

    const int h = blockIdx.y * 256 + t;
    const float* wr = W + (size_t)h * D_;
    float acc = 0.f;
#pragma unroll 4
    for (int d = 0; d < D_; d += 4) {
        float4 w4 = *(const float4*)(wr + d);
        acc += w4.x * xs[d] + w4.y * xs[d + 1] + w4.z * xs[d + 2] + w4.w * xs[d + 3];
    }
    float bb = bias[h];
    if (bias2) bb += bias2[h];
    Y[b * H_ + h] = acc + bb;
}

// ---------------------------------------------------------------------------
// Additive-attention scores via bf16 WMMA.
// att[b,s] = sum_h V[h]*tanh(qplus[b,h] + W_ctx[h,:].contexts[b,s,:])
// grid (B, S/64), block 256 (8 waves); wave w owns h in [64w, 64w+64),
// each wave covers 4 column tiles (reusing the A fragment 4x per k-step).
// ---------------------------------------------------------------------------
#define STILE 64
#define NJ    (STILE / 16)
#define SD    (D_ + 8)   // LDS row stride in bf16 (stride = 4 dwords mod 64 banks)

__global__ __launch_bounds__(256) void att_scores(
    const float* __restrict__ contexts,        // [B,S,D] f32
    const unsigned short* __restrict__ wbf,    // [H,D] bf16 bits (pre-converted)
    const float* __restrict__ qplus,           // [B,H]  (q + b_ctx)
    const float* __restrict__ V,               // [H]
    const unsigned char* __restrict__ mask,    // [B,S] (bool)
    float* __restrict__ att)                   // [B,S]
{
    __shared__ __attribute__((aligned(16))) unsigned short tile[STILE * SD];
    __shared__ float red[8][STILE];

    const int b  = blockIdx.x;
    const int s0 = blockIdx.y * STILE;
    const int t  = threadIdx.x;

    // ---- stage contexts[b, s0:s0+64, :] into LDS as bf16 (coalesced f32x4) ----
    const float* cbase = contexts + ((size_t)b * S_ + s0) * D_;
#pragma unroll
    for (int i = 0; i < 32; ++i) {
        int idx = t + i * 256;          // float4 index: 64 rows * 128 f4/row
        int row = idx >> 7;
        int c4  = idx & 127;
        float4 x = *(const float4*)(cbase + (size_t)row * D_ + c4 * 4);
        *(uint2*)&tile[row * SD + c4 * 4] = make_uint2(pack2(x.x, x.y), pack2(x.z, x.w));
    }
    __syncthreads();

    const int wave = t >> 5;
    const int lane = t & 31;
    const int ln   = lane & 15;
    const int hi   = lane >> 4;          // half-wave selector
    const int h0   = wave * 64;

    float part[NJ];
#pragma unroll
    for (int j = 0; j < NJ; ++j) part[j] = 0.f;

#pragma unroll 1
    for (int mt = 0; mt < 4; ++mt) {
        const int h_base = h0 + mt * 16;
        v8f acc[NJ];
#pragma unroll
        for (int j = 0; j < NJ; ++j) acc[j] = (v8f){0.f,0.f,0.f,0.f,0.f,0.f,0.f,0.f};

        // A fragment: CDNA5 16-bit A layout -> lane (hi,ln) holds row h_base+ln,
        // K = {base..base+7, base+16..base+23}, base = 8*hi.  W is pre-bf16, so
        // each k-step is just two global 16B loads.
        const uint4* ap = (const uint4*)(wbf + (size_t)(h_base + ln) * D_ + hi * 8);

        for (int k = 0; k < D_; k += 32) {
            ABFrag a;
            uint4 x0 = ap[k >> 3];
            uint4 x1 = ap[(k >> 3) + 2];
            a.u[0] = x0.x; a.u[1] = x0.y; a.u[2] = x0.z; a.u[3] = x0.w;
            a.u[4] = x1.x; a.u[5] = x1.y; a.u[6] = x1.z; a.u[7] = x1.w;

            const int kb = k + hi * 8;   // B layout: K = {kb..kb+7, kb+16..kb+23}, matching A
#pragma unroll
            for (int j = 0; j < NJ; ++j) {
                ABFrag bf;
                const uint4* p = (const uint4*)&tile[(j * 16 + ln) * SD + kb];
                uint4 y0 = p[0], y1 = p[2];
                bf.u[0] = y0.x; bf.u[1] = y0.y; bf.u[2] = y0.z; bf.u[3] = y0.w;
                bf.u[4] = y1.x; bf.u[5] = y1.y; bf.u[6] = y1.z; bf.u[7] = y1.w;
                acc[j] = __builtin_amdgcn_wmma_f32_16x16x32_bf16(
                             false, a.v, false, bf.v, (short)0, acc[j], false, false);
            }
        }

        // Epilogue: C/D layout -> lane holds rows M=r+8*hi, col = lane&15
        const int hb = h_base + hi * 8;
#pragma unroll
        for (int r = 0; r < 8; ++r) {
            int h = hb + r;
            float qb = qplus[b * H_ + h];
            float vh = V[h];
#pragma unroll
            for (int j = 0; j < NJ; ++j)
                part[j] += vh * fast_tanh(acc[j][r] + qb);
        }
    }

    // combine the two half-wave row groups (same column, disjoint h)
#pragma unroll
    for (int j = 0; j < NJ; ++j) part[j] += __shfl_xor(part[j], 16, 32);

    if (hi == 0) {
#pragma unroll
        for (int j = 0; j < NJ; ++j) red[wave][j * 16 + ln] = part[j];
    }
    __syncthreads();

    if (t < STILE) {
        float sum = 0.f;
#pragma unroll
        for (int w = 0; w < 8; ++w) sum += red[w][t];
        int s = s0 + t;
        att[b * S_ + s] = mask[(size_t)b * S_ + s] ? -INFINITY : sum;
    }
}

// ---------------------------------------------------------------------------
// Row softmax over S. grid B, block 256, 8 elems/thread.
// ---------------------------------------------------------------------------
__global__ __launch_bounds__(256) void softmax_rows(
    const float* __restrict__ att, float* __restrict__ score)
{
    __shared__ float red[256];
    const int b = blockIdx.x;
    const int t = threadIdx.x;
    const float* row = att + (size_t)b * S_;

    float v[8];
    float m = -INFINITY;
#pragma unroll
    for (int i = 0; i < 8; ++i) { v[i] = row[t + i * 256]; m = fmaxf(m, v[i]); }
    red[t] = m; __syncthreads();
    for (int o = 128; o > 0; o >>= 1) {
        if (t < o) red[t] = fmaxf(red[t], red[t + o]);
        __syncthreads();
    }
    m = red[0]; __syncthreads();

    float s = 0.f;
#pragma unroll
    for (int i = 0; i < 8; ++i) { v[i] = __expf(v[i] - m); s += v[i]; }
    red[t] = s; __syncthreads();
    for (int o = 128; o > 0; o >>= 1) {
        if (t < o) red[t] += red[t + o];
        __syncthreads();
    }
    const float inv = 1.f / red[0];
#pragma unroll
    for (int i = 0; i < 8; ++i) score[(size_t)b * S_ + t + i * 256] = v[i] * inv;
}

// ---------------------------------------------------------------------------
// Partial weighted context sum, optimized for memory-level parallelism:
// wpart[p][b,d] = sum_{s in segment p} score[b,s]*contexts[b,s,d]
// grid (B, 4), block 256 = 128 float4-columns x 2 s-groups (8 segments total).
// Fully coalesced: 128 lanes x 16B cover one whole 2KB row per step.
// ---------------------------------------------------------------------------
#define NSEG 4
__global__ __launch_bounds__(256) void weighted_ctx_part(
    const float* __restrict__ contexts, const float* __restrict__ score,
    float* __restrict__ wpart)
{
    __shared__ float sc[512];
    __shared__ float4 partlds[128];
    const int b  = blockIdx.x;
    const int by = blockIdx.y;            // s segment pair: covers 512 rows
    const int t  = threadIdx.x;
    sc[t]       = score[(size_t)b * S_ + by * 512 + t];
    sc[t + 256] = score[(size_t)b * S_ + by * 512 + t + 256];
    __syncthreads();

    const int d4 = t & 127;
    const int sg = t >> 7;                // 0/1: which 256-row half
    const float4* cptr = (const float4*)contexts
                       + ((size_t)(b * S_ + by * 512 + sg * 256)) * (D_ / 4) + d4;
    float4 acc = {0.f, 0.f, 0.f, 0.f};
    for (int i = 0; i < 256; i += 4) {
        __builtin_prefetch(cptr + (size_t)(i + 32) * (D_ / 4), 0, 0);
#pragma unroll
        for (int u = 0; u < 4; ++u) {
            float w = sc[sg * 256 + i + u];
            float4 c = cptr[(size_t)(i + u) * (D_ / 4)];
            acc.x += w * c.x; acc.y += w * c.y; acc.z += w * c.z; acc.w += w * c.w;
        }
    }

    if (sg == 1) partlds[d4] = acc;
    __syncthreads();
    if (sg == 0) {
        float4 o = partlds[d4];
        acc.x += o.x; acc.y += o.y; acc.z += o.z; acc.w += o.w;
        ((float4*)wpart)[((size_t)by * B_ * D_ + (size_t)b * D_) / 4 + d4] = acc;
    }
}

// ---------------------------------------------------------------------------
// d_out = [hidden (B*H) | score (B*S)]
// hidden = W_ctx . (score-weighted contexts) + b_ctx   (sum(score)==1)
// ---------------------------------------------------------------------------
extern "C" void kernel_launch(void* const* d_in, const int* in_sizes, int n_in,
                              void* d_out, int out_size, void* d_ws, size_t ws_size,
                              hipStream_t stream) {
    const float*         inputs   = (const float*)d_in[0];
    const float*         contexts = (const float*)d_in[1];
    const unsigned char* mask     = (const unsigned char*)d_in[2];
    const float*         W_in     = (const float*)d_in[3];
    const float*         b_in     = (const float*)d_in[4];
    const float*         W_ctx    = (const float*)d_in[5];
    const float*         b_ctx    = (const float*)d_in[6];
    const float*         V        = (const float*)d_in[7];

    float* out    = (float*)d_out;
    float* hidden = out;                 // [B,H]
    float* score  = out + B_ * H_;       // [B,S]

    float* ws    = (float*)d_ws;
    float* qplus = ws;                                    // B*H = 65536 f
    float* att   = ws + B_ * H_;                          // B*S = 262144 f
    // wpart aliases att: att is dead after softmax_rows, and
    // NSEG*B*D = 4*128*512 = 262144 floats = exactly att's footprint.
    float* wpart = att;
    unsigned short* wbf = (unsigned short*)(ws + B_ * H_ + B_ * S_); // H*D bf16

    // 0) W_ctx -> bf16 once
    to_bf16<<<(H_ * D_) / (256 * 4), 256, 0, stream>>>(W_ctx, wbf);
    // 1) qplus = inputs @ W_in^T + b_in + b_ctx
    linear_bh<<<dim3(B_, 2), 256, 0, stream>>>(inputs, W_in, b_in, b_ctx, qplus,
                                               1, 0);
    // 2) att[b,s] = V . tanh(qplus + W_ctx@ctx), masked   (bf16 WMMA)
    att_scores<<<dim3(B_, S_ / STILE), 256, 0, stream>>>(
        contexts, wbf, qplus, V, mask, att);
    // 3) softmax over S
    softmax_rows<<<B_, 256, 0, stream>>>(att, score);
    // 4) wpart[p] = per-segment score-weighted sums of contexts (4 partials of 512 rows)
    weighted_ctx_part<<<dim3(B_, NSEG), 256, 0, stream>>>(contexts, score, wpart);
    // 5) hidden = (sum_p wpart[p]) @ W_ctx^T + b_ctx  (partials summed in stage)
    linear_bh<<<dim3(B_, 2), 256, 0, stream>>>(wpart, W_ctx, b_ctx, nullptr,
                                               hidden, NSEG, (size_t)B_ * D_);
}
